// Enhanced_Physics_Attention_with_Sonata_24953759990234
// MI455X (gfx1250) — matrix operand order repaired and checked by reading, not compiled, not run on hardware
//
#include <hip/hip_runtime.h>
#include <hip/hip_bf16.h>
#include <cstdint>

#define B_   4
#define N_   16384
#define C_   256
#define H_   8
#define D_   64
#define G_   64
#define M_   1024
#define SD_  768
#define HD_  512
#define SCALE_ 0.125f

typedef __attribute__((ext_vector_type(16))) __bf16 v16bf;
typedef __attribute__((ext_vector_type(8)))  __bf16 v8bf;
typedef __attribute__((ext_vector_type(8)))  float  v8f;

typedef __attribute__((ext_vector_type(4))) unsigned int tdm_u32x4;
typedef __attribute__((ext_vector_type(8))) int          tdm_i32x8;
typedef __attribute__((ext_vector_type(4))) int          tdm_i32x4;

__device__ __forceinline__ v8f vzero8() {
  v8f v;
#pragma unroll
  for (int i = 0; i < 8; ++i) v[i] = 0.0f;
  return v;
}

__device__ __forceinline__ v8f wmma_bf16(v16bf a, v16bf b, v8f c) {
  return __builtin_amdgcn_wmma_f32_16x16x32_bf16(false, a, false, b, (short)0, c, false, false);
}

// ---- TDM: async 2D bf16 tile load global->LDS with padded LDS pitch ----
// Copies tile1 rows x tile0 bf16 elems; global row stride = stride0 elems;
// LDS pitch = tile0*2 + 16 bytes (pad 4 DWORDs every 32 DWORDs => 64->72 elem pitch).
__device__ __forceinline__ void tdm_load_tile_bf16(
    unsigned lds_addr, const __bf16* gptr,
    unsigned tile0, unsigned tile1,
    unsigned tdim0, unsigned tdim1, unsigned stride0_elems)
{
  unsigned long long ga = (unsigned long long)gptr;
  tdm_u32x4 g0 = {
      1u,                                            // count=1, user descriptor
      lds_addr,                                      // lds byte address
      (unsigned)(ga & 0xffffffffu),                  // global addr [31:0]
      (unsigned)((ga >> 32) & 0x01ffffffu) | (2u << 30)  // addr[56:32], type=2
  };
  tdm_i32x8 g1 = {
      (int)((1u << 16) | (1u << 20) | (4u << 22) | (3u << 25)),
      //    data_size=2B  pad_enable  interval=32DW  amount=4DW
      (int)((tdim0 & 0xffffu) << 16),                          // tensor_dim0[15:0]
      (int)((tdim0 >> 16) | ((tdim1 & 0xffffu) << 16)),        // td0[31:16], td1[15:0]
      (int)((tdim1 >> 16) | ((tile0 & 0xffffu) << 16)),        // td1[31:16], tile_dim0
      (int)(tile1 & 0xffffu),                                  // tile_dim1 (tile_dim2=0)
      (int)stride0_elems,                                      // tensor_dim0_stride[31:0]
      0, 0
  };
  tdm_i32x4 z4 = {0, 0, 0, 0};
#if defined(__clang_major__) && (__clang_major__ >= 23)
  tdm_i32x8 z8 = {0, 0, 0, 0, 0, 0, 0, 0};
  __builtin_amdgcn_tensor_load_to_lds(g0, g1, z4, z4, z8, 0);
#else
  __builtin_amdgcn_tensor_load_to_lds(g0, g1, z4, z4, 0);
#endif
}

__device__ __forceinline__ unsigned lds_off_of(const void* p) {
  return (unsigned)(unsigned long long)p;   // flat->LDS: low 32 bits are LDS byte addr
}

// A fragment: A is 16(M) x 32(K), row-major, lda elements.
__device__ __forceinline__ v16bf frag_a(const __bf16* base, int lda) {
  const int l = threadIdx.x & 31;
  const __bf16* p = base + (l & 15) * lda + ((l >> 4) << 3);
  v8bf lo = *(const v8bf*)p;
  v8bf hi = *(const v8bf*)(p + 16);
  v16bf f;
#pragma unroll
  for (int i = 0; i < 8; ++i) { f[i] = lo[i]; f[8 + i] = hi[i]; }
  return f;
}

// B fragment: B is 32(K) x 16(N); baseT is the TRANSPOSED buffer [N][K] row-major.
__device__ __forceinline__ v16bf frag_b(const __bf16* baseT, int ldb) {
  const int l = threadIdx.x & 31;
  const __bf16* p = baseT + (l & 15) * ldb + ((l >> 4) << 3);
  v8bf a0 = *(const v8bf*)p;
  v8bf a1 = *(const v8bf*)(p + 16);
  v16bf f;
#pragma unroll
  for (int i = 0; i < 8; ++i) { f[i] = a0[i]; f[8 + i] = a1[i]; }
  return f;
}

// C/D layout: lane l col=l&15, VGPR i row = (l>>4)*8 + i
__device__ __forceinline__ void store_c_f32(float* base, int ldc, v8f acc) {
  const int l = threadIdx.x & 31;
  const int c = l & 15, r0 = (l >> 4) << 3;
#pragma unroll
  for (int i = 0; i < 8; ++i) base[(r0 + i) * ldc + c] = acc[i];
}

__device__ __forceinline__ void store_c_bf16(__bf16* base, long long ldc, v8f acc) {
  const int l = threadIdx.x & 31;
  const int c = l & 15, r0 = (l >> 4) << 3;
#pragma unroll
  for (int i = 0; i < 8; ++i) base[(long long)(r0 + i) * ldc + c] = (__bf16)acc[i];
}

__device__ __forceinline__ void store_c_bf16_T(__bf16* base, int ldc, v8f acc) {
  const int l = threadIdx.x & 31;
  const int c = l & 15, r0 = (l >> 4) << 3;
#pragma unroll
  for (int i = 0; i < 8; ++i) base[c * ldc + r0 + i] = (__bf16)acc[i];
}

// ---------------- prep ----------------
__global__ void k_transpose_bf16(const float* __restrict__ W, __bf16* __restrict__ WT,
                                 int K, int Nn) {
  int idx = blockIdx.x * blockDim.x + threadIdx.x;
  if (idx >= K * Nn) return;
  int n = idx / K, k = idx - n * K;
  WT[idx] = (__bf16)W[k * Nn + n];
}

__global__ void k_zero_f32(float* __restrict__ p, int n) {
  int i = blockIdx.x * blockDim.x + threadIdx.x;
  if (i < n) p[i] = 0.0f;
}

// ---------------- K1: projections + slice softmax ----------------
__global__ __launch_bounds__(128) void k_proj(
    const float* __restrict__ x, const float* __restrict__ temperature,
    const __bf16* __restrict__ WxT, const float* __restrict__ bx,
    const __bf16* __restrict__ WfxT, const float* __restrict__ bfx,
    const __bf16* __restrict__ WslT, const float* __restrict__ bsl,
    __bf16* __restrict__ fx_ws, __bf16* __restrict__ w_ws)
{
  __shared__ __bf16 xt[64 * 264];
  __shared__ __bf16 xm[64 * 72];
  __shared__ float  sc[64 * 68];
  const int tid = threadIdx.x, lane = tid & 31, wave = tid >> 5;
  const int g0 = wave * 16;
  const long long tile0 = (long long)blockIdx.x * 64;
  const int b = (int)(tile0 / N_);
  const int nloc0 = (int)(tile0 % N_);

  const float4* xv = (const float4*)(x + tile0 * C_);
  for (int idx = tid; idx < 64 * (C_ / 4); idx += 128) {
    float4 v = xv[idx];
    int r = idx >> 6, c4 = (idx & 63) << 2;
    __bf16* d = xt + r * 264 + c4;
    d[0] = (__bf16)v.x; d[1] = (__bf16)v.y; d[2] = (__bf16)v.z; d[3] = (__bf16)v.w;
  }
  __syncthreads();

  for (int h = 0; h < H_; ++h) {
    v8f acc[4];
#pragma unroll
    for (int t = 0; t < 4; ++t) acc[t] = vzero8();
#pragma unroll
    for (int kk = 0; kk < 8; ++kk) {
      v16bf a = frag_a(xt + g0 * 264 + kk * 32, 264);
#pragma unroll
      for (int t = 0; t < 4; ++t)
        acc[t] = wmma_bf16(a, frag_b(WxT + (h * 64 + t * 16) * C_ + kk * 32, C_), acc[t]);
    }
#pragma unroll
    for (int t = 0; t < 4; ++t) {
      float bb = bx[h * 64 + t * 16 + (lane & 15)];
#pragma unroll
      for (int i = 0; i < 8; ++i) acc[t][i] += bb;
      store_c_bf16(xm + g0 * 72 + t * 16, 72, acc[t]);
    }
    // logits = xm @ WslT  (same-wave stripe)
    v8f lac[4];
#pragma unroll
    for (int t = 0; t < 4; ++t) lac[t] = vzero8();
#pragma unroll
    for (int kk = 0; kk < 2; ++kk) {
      v16bf a = frag_a(xm + g0 * 72 + kk * 32, 72);
#pragma unroll
      for (int t = 0; t < 4; ++t)
        lac[t] = wmma_bf16(a, frag_b(WslT + (t * 16) * 64 + kk * 32, 64), lac[t]);
    }
#pragma unroll
    for (int t = 0; t < 4; ++t) {
      float bb = bsl[t * 16 + (lane & 15)];
#pragma unroll
      for (int i = 0; i < 8; ++i) lac[t][i] += bb;
      store_c_f32(sc + g0 * 68 + t * 16, 68, lac[t]);
    }
    __syncthreads();
    if (tid < 64) {   // row softmax with temperature
      float invt = 1.0f / temperature[h];
      float* row = sc + tid * 68;
      float mx = -1e30f;
      for (int g2 = 0; g2 < 64; ++g2) { float s = row[g2] * invt; row[g2] = s; mx = fmaxf(mx, s); }
      float sum = 0.f;
      for (int g2 = 0; g2 < 64; ++g2) { float e = __expf(row[g2] - mx); row[g2] = e; sum += e; }
      float inv = 1.0f / sum;
      __bf16* wdst = w_ws + (((long long)(b * H_ + h)) * N_ + nloc0 + tid) * G_;
      for (int g2 = 0; g2 < 64; ++g2) wdst[g2] = (__bf16)(row[g2] * inv);
    }
    __syncthreads();
    // fx stripe
    v8f fac[4];
#pragma unroll
    for (int t = 0; t < 4; ++t) fac[t] = vzero8();
#pragma unroll
    for (int kk = 0; kk < 8; ++kk) {
      v16bf a = frag_a(xt + g0 * 264 + kk * 32, 264);
#pragma unroll
      for (int t = 0; t < 4; ++t)
        fac[t] = wmma_bf16(a, frag_b(WfxT + (h * 64 + t * 16) * C_ + kk * 32, C_), fac[t]);
    }
#pragma unroll
    for (int t = 0; t < 4; ++t) {
      float bb = bfx[h * 64 + t * 16 + (lane & 15)];
#pragma unroll
      for (int i = 0; i < 8; ++i) fac[t][i] += bb;
      store_c_bf16(fx_ws + (tile0 + g0) * HD_ + h * 64 + t * 16, HD_, fac[t]);
    }
  }
}

// ---------------- K2: weighted pooling (slice tokens) ----------------
__global__ __launch_bounds__(128) void k_pool(
    const __bf16* __restrict__ w_ws, const __bf16* __restrict__ fx_ws,
    float* __restrict__ stok, float* __restrict__ snorm)
{
  __shared__ __bf16 wt[64 * 72];   // [g][n] (transposed stage, TDM can't transpose)
  __shared__ __bf16 fxt[64 * 72];  // [d][n]
  const int tid = threadIdx.x, lane = tid & 31, wave = tid >> 5;
  const int g0 = wave * 16;
  const int bh = blockIdx.x, b = bh / H_, h = bh - b * H_;
  const int nbase = blockIdx.y * (N_ / 8);
  v8f acc[4];
#pragma unroll
  for (int t = 0; t < 4; ++t) acc[t] = vzero8();
  float normacc = 0.f;

  for (int chunk = 0; chunk < (N_ / 8) / 64; ++chunk) {
    const int n0 = nbase + chunk * 64;
    for (int idx = tid; idx < 4096; idx += 128) {
      int nl = idx >> 6, c = idx & 63;
      wt[c * 72 + nl]  = w_ws[((long long)bh * N_ + n0 + nl) * G_ + c];
      fxt[c * 72 + nl] = fx_ws[((long long)(b * N_ + n0 + nl)) * HD_ + h * 64 + c];
    }
    __syncthreads();
    if (tid < 64) {
      float s = 0.f;
      for (int nl = 0; nl < 64; ++nl) s += (float)wt[tid * 72 + nl];
      normacc += s;
    }
#pragma unroll
    for (int kk = 0; kk < 2; ++kk) {
      v16bf a = frag_a(wt + g0 * 72 + kk * 32, 72);
#pragma unroll
      for (int t = 0; t < 4; ++t)
        acc[t] = wmma_bf16(a, frag_b(fxt + (t * 16) * 72 + kk * 32, 72), acc[t]);
    }
    __syncthreads();
  }
#pragma unroll
  for (int t = 0; t < 4; ++t) {
    int cc = t * 16 + (lane & 15);
#pragma unroll
    for (int i = 0; i < 8; ++i) {
      int rr = g0 + ((lane >> 4) << 3) + i;
      atomicAdd(&stok[((long long)bh * G_ + rr) * D_ + cc], acc[t][i]);
    }
  }
  if (tid < 64) atomicAdd(&snorm[bh * G_ + tid], normacc);
}

// ---------------- KS: sonata projection ----------------
__global__ __launch_bounds__(128) void k_sonata(
    const float* __restrict__ sfeat, const __bf16* __restrict__ WspT,
    const float* __restrict__ bsp, __bf16* __restrict__ sf_ws)
{
  __shared__ __bf16 xt[64 * 776];
  const int tid = threadIdx.x, lane = tid & 31, wave = tid >> 5;
  const int g0 = wave * 16;
  const long long tile0 = (long long)blockIdx.x * 64;

  const float4* xv = (const float4*)(sfeat + tile0 * SD_);
  for (int idx = tid; idx < 64 * (SD_ / 4); idx += 128) {
    float4 v = xv[idx];
    int r = idx / (SD_ / 4), c4 = (idx % (SD_ / 4)) << 2;
    __bf16* d = xt + r * 776 + c4;
    d[0] = (__bf16)v.x; d[1] = (__bf16)v.y; d[2] = (__bf16)v.z; d[3] = (__bf16)v.w;
  }
  __syncthreads();

  for (int cg = 0; cg < 8; ++cg) {
    v8f acc[4];
#pragma unroll
    for (int t = 0; t < 4; ++t) acc[t] = vzero8();
    for (int kk = 0; kk < SD_ / 32; ++kk) {
      v16bf a = frag_a(xt + g0 * 776 + kk * 32, 776);
#pragma unroll
      for (int t = 0; t < 4; ++t)
        acc[t] = wmma_bf16(a, frag_b(WspT + (cg * 64 + t * 16) * SD_ + kk * 32, SD_), acc[t]);
    }
#pragma unroll
    for (int t = 0; t < 4; ++t) {
      float bb = bsp[cg * 64 + t * 16 + (lane & 15)];
#pragma unroll
      for (int i = 0; i < 8; ++i) acc[t][i] += bb;
      store_c_bf16(sf_ws + (tile0 + g0) * HD_ + cg * 64 + t * 16, HD_, acc[t]);
    }
  }
}

// ---------------- K3: slice self-attn + sonata cross-attn ----------------
__global__ __launch_bounds__(128) void k_attn(
    const float* __restrict__ stok, const float* __restrict__ snorm,
    const __bf16* __restrict__ sf_ws,
    const __bf16* __restrict__ WqT, const float* __restrict__ bq,
    const __bf16* __restrict__ WkT, const float* __restrict__ bk,
    const __bf16* __restrict__ WvT, const float* __restrict__ bv,
    const __bf16* __restrict__ WckT, const float* __restrict__ bck,
    const __bf16* __restrict__ WcvT, const float* __restrict__ bcv,
    __bf16* __restrict__ osT)
{
  __shared__ __bf16 st[64 * 72];    // slice tokens [g][d]
  __shared__ __bf16 qs[64 * 72];    // q, later o1 [g][d]
  __shared__ __bf16 ks[64 * 72];    // k [g][d]
  __shared__ __bf16 vT[64 * 72];    // v transposed [d][g]
  __shared__ __bf16 pb[64 * 72];    // probs bf16
  __shared__ __bf16 sfc[64 * 72];   // sf chunk [m][d]  (TDM target)
  __shared__ __bf16 ksc[64 * 72];   // k_s chunk [m][d]
  __shared__ __bf16 vsT[64 * 72];   // v_s^T [d][m]
  __shared__ float  sc[64 * 68];
  __shared__ float  rm[64], rl[64], rfac[64];

  const int tid = threadIdx.x, lane = tid & 31, wave = tid >> 5;
  const int g0 = wave * 16;
  const int bh = blockIdx.x, b = bh / H_, h = bh - b * H_;

  for (int idx = tid; idx < 4096; idx += 128) {
    int g2 = idx >> 6;
    float v = stok[(long long)bh * 4096 + idx] / (snorm[bh * 64 + g2] + 1e-5f);
    st[g2 * 72 + (idx & 63)] = (__bf16)v;
  }
  if (tid < 64) { rm[tid] = -1e30f; rl[tid] = 0.f; }
  __syncthreads();

  { // q,k,v
    v8f qa[4], ka[4], va[4];
#pragma unroll
    for (int t = 0; t < 4; ++t) { qa[t] = vzero8(); ka[t] = vzero8(); va[t] = vzero8(); }
#pragma unroll
    for (int kk = 0; kk < 2; ++kk) {
      v16bf a = frag_a(st + g0 * 72 + kk * 32, 72);
#pragma unroll
      for (int t = 0; t < 4; ++t) {
        qa[t] = wmma_bf16(a, frag_b(WqT + (t * 16) * 64 + kk * 32, 64), qa[t]);
        ka[t] = wmma_bf16(a, frag_b(WkT + (t * 16) * 64 + kk * 32, 64), ka[t]);
        va[t] = wmma_bf16(a, frag_b(WvT + (t * 16) * 64 + kk * 32, 64), va[t]);
      }
    }
#pragma unroll
    for (int t = 0; t < 4; ++t) {
      float b0 = bq[t * 16 + (lane & 15)], b1 = bk[t * 16 + (lane & 15)], b2 = bv[t * 16 + (lane & 15)];
#pragma unroll
      for (int i = 0; i < 8; ++i) { qa[t][i] += b0; ka[t][i] += b1; va[t][i] += b2; }
      store_c_bf16(qs + g0 * 72 + t * 16, 72, qa[t]);
      store_c_bf16(ks + g0 * 72 + t * 16, 72, ka[t]);
      store_c_bf16_T(vT + (t * 16) * 72 + g0, 72, va[t]);
    }
  }
  __syncthreads();
  { // scores = q @ k^T * scale
    v8f sa[4];
#pragma unroll
    for (int t = 0; t < 4; ++t) sa[t] = vzero8();
#pragma unroll
    for (int kk = 0; kk < 2; ++kk) {
      v16bf a = frag_a(qs + g0 * 72 + kk * 32, 72);
#pragma unroll
      for (int t = 0; t < 4; ++t)
        sa[t] = wmma_bf16(a, frag_b(ks + (t * 16) * 72 + kk * 32, 72), sa[t]);
    }
#pragma unroll
    for (int t = 0; t < 4; ++t) {
#pragma unroll
      for (int i = 0; i < 8; ++i) sa[t][i] *= SCALE_;
      store_c_f32(sc + g0 * 68 + t * 16, 68, sa[t]);
    }
  }
  __syncthreads();
  if (tid < 64) { // softmax
    float* row = sc + tid * 68;
    float mx = -1e30f;
    for (int g2 = 0; g2 < 64; ++g2) mx = fmaxf(mx, row[g2]);
    float s = 0.f;
    for (int g2 = 0; g2 < 64; ++g2) { float e = __expf(row[g2] - mx); s += e; row[g2] = e; }
    float inv = 1.0f / s;
    for (int g2 = 0; g2 < 64; ++g2) pb[tid * 72 + g2] = (__bf16)(row[g2] * inv);
  }
  __syncthreads();
  // o1 = probs @ v
  v8f o1[4];
#pragma unroll
  for (int t = 0; t < 4; ++t) o1[t] = vzero8();
#pragma unroll
  for (int kk = 0; kk < 2; ++kk) {
    v16bf a = frag_a(pb + g0 * 72 + kk * 32, 72);
#pragma unroll
    for (int t = 0; t < 4; ++t)
      o1[t] = wmma_bf16(a, frag_b(vT + (t * 16) * 72 + kk * 32, 72), o1[t]);
  }
  __syncthreads();
#pragma unroll
  for (int t = 0; t < 4; ++t) store_c_bf16(qs + g0 * 72 + t * 16, 72, o1[t]); // qs := o1
  __syncthreads();

  // cross-attention over sonata tokens (flash-style), sf chunk staged by TDM
  v8f racc[4];
#pragma unroll
  for (int t = 0; t < 4; ++t) racc[t] = vzero8();
  for (int mc = 0; mc < M_ / 64; ++mc) {
    const int m0 = mc * 64;
    if (tid < 32) {   // wave 0 drives the Tensor Data Mover
      tdm_load_tile_bf16(lds_off_of(sfc),
                         sf_ws + ((long long)(b * M_ + m0)) * HD_ + h * 64,
                         64, 64, 64, 64, HD_);
      __builtin_amdgcn_s_wait_tensorcnt(0);
    }
    __syncthreads();
    { // k_s, v_s for this chunk (wave stripe over m)
      v8f kca[4], vca[4];
#pragma unroll
      for (int t = 0; t < 4; ++t) { kca[t] = vzero8(); vca[t] = vzero8(); }
#pragma unroll
      for (int kk = 0; kk < 2; ++kk) {
        v16bf a = frag_a(sfc + g0 * 72 + kk * 32, 72);
#pragma unroll
        for (int t = 0; t < 4; ++t) {
          kca[t] = wmma_bf16(a, frag_b(WckT + (t * 16) * 64 + kk * 32, 64), kca[t]);
          vca[t] = wmma_bf16(a, frag_b(WcvT + (t * 16) * 64 + kk * 32, 64), vca[t]);
        }
      }
#pragma unroll
      for (int t = 0; t < 4; ++t) {
        float b0 = bck[t * 16 + (lane & 15)], b1 = bcv[t * 16 + (lane & 15)];
#pragma unroll
        for (int i = 0; i < 8; ++i) { kca[t][i] += b0; vca[t][i] += b1; }
        store_c_bf16(ksc + g0 * 72 + t * 16, 72, kca[t]);
        store_c_bf16_T(vsT + (t * 16) * 72 + g0, 72, vca[t]);
      }
    }
    __syncthreads();
    { // cs = o1 @ k_s^T * scale
      v8f ca[4];
#pragma unroll
      for (int t = 0; t < 4; ++t) ca[t] = vzero8();
#pragma unroll
      for (int kk = 0; kk < 2; ++kk) {
        v16bf a = frag_a(qs + g0 * 72 + kk * 32, 72);
#pragma unroll
        for (int t = 0; t < 4; ++t)
          ca[t] = wmma_bf16(a, frag_b(ksc + (t * 16) * 72 + kk * 32, 72), ca[t]);
      }
#pragma unroll
      for (int t = 0; t < 4; ++t) {
#pragma unroll
        for (int i = 0; i < 8; ++i) ca[t][i] *= SCALE_;
        store_c_f32(sc + g0 * 68 + t * 16, 68, ca[t]);
      }
    }
    __syncthreads();
    if (tid < 64) { // online softmax update
      float* row = sc + tid * 68;
      float mx = rm[tid];
      for (int m2 = 0; m2 < 64; ++m2) mx = fmaxf(mx, row[m2]);
      float fac = __expf(rm[tid] - mx);
      float s = 0.f;
      for (int m2 = 0; m2 < 64; ++m2) {
        float e = __expf(row[m2] - mx);
        s += e;
        pb[tid * 72 + m2] = (__bf16)e;
      }
      rl[tid] = rl[tid] * fac + s;
      rm[tid] = mx;
      rfac[tid] = fac;
    }
    __syncthreads();
#pragma unroll
    for (int t = 0; t < 4; ++t) {
#pragma unroll
      for (int i = 0; i < 8; ++i) {
        int rr = g0 + ((lane >> 4) << 3) + i;
        racc[t][i] *= rfac[rr];
      }
    }
#pragma unroll
    for (int kk = 0; kk < 2; ++kk) {
      v16bf a = frag_a(pb + g0 * 72 + kk * 32, 72);
#pragma unroll
      for (int t = 0; t < 4; ++t)
        racc[t] = wmma_bf16(a, frag_b(vsT + (t * 16) * 72 + kk * 32, 72), racc[t]);
    }
    __syncthreads();
  }
  // final: out_slice = o1 + racc / l ; store TRANSPOSED [d][g] to global
#pragma unroll
  for (int t = 0; t < 4; ++t) {
    int cc = t * 16 + (lane & 15);
#pragma unroll
    for (int i = 0; i < 8; ++i) {
      int rr = g0 + ((lane >> 4) << 3) + i;
      float val = o1[t][i] + racc[t][i] / rl[rr];
      osT[(long long)bh * 4096 + cc * 64 + rr] = (__bf16)val;
    }
  }
}

// ---------------- K4: unpool + output projection ----------------
__global__ __launch_bounds__(128) void k_unpool_out(
    const __bf16* __restrict__ w_ws, const __bf16* __restrict__ osT,
    const __bf16* __restrict__ WoT, const float* __restrict__ bo,
    float* __restrict__ out)
{
  __shared__ __bf16 wl[64 * 72];   // [n][g]  (TDM target)
  __shared__ __bf16 osl[64 * 72];  // [d][g]  (TDM target)
  __shared__ __bf16 ox[64 * 520];  // [n][512]
  const int tid = threadIdx.x, lane = tid & 31, wave = tid >> 5;
  const int r0w = wave * 16;
  const long long tile0 = (long long)blockIdx.x * 64;
  const int b = (int)(tile0 / N_);
  const int nloc0 = (int)(tile0 % N_);

  for (int h = 0; h < H_; ++h) {
    const int bh = b * H_ + h;
    if (tid < 32) {   // wave 0 drives the TDM for both tiles
      tdm_load_tile_bf16(lds_off_of(wl),
                         w_ws + ((long long)bh * N_ + nloc0) * G_,
                         64, 64, 64, 64, G_);
      tdm_load_tile_bf16(lds_off_of(osl),
                         osT + (long long)bh * 4096,
                         64, 64, 64, 64, 64);
      __builtin_amdgcn_s_wait_tensorcnt(0);
    }
    __syncthreads();
    v8f acc[4];
#pragma unroll
    for (int t = 0; t < 4; ++t) acc[t] = vzero8();
#pragma unroll
    for (int kk = 0; kk < 2; ++kk) {
      v16bf a = frag_a(wl + r0w * 72 + kk * 32, 72);
#pragma unroll
      for (int t = 0; t < 4; ++t)
        acc[t] = wmma_bf16(a, frag_b(osl + (t * 16) * 72 + kk * 32, 72), acc[t]);
    }
#pragma unroll
    for (int t = 0; t < 4; ++t)
      store_c_bf16(ox + r0w * 520 + h * 64 + t * 16, 520, acc[t]);
    __syncthreads();
  }
  // out = ox @ WoT + bo
  for (int cg = 0; cg < 4; ++cg) {
    v8f acc[4];
#pragma unroll
    for (int t = 0; t < 4; ++t) acc[t] = vzero8();
    for (int kk = 0; kk < HD_ / 32; ++kk) {
      v16bf a = frag_a(ox + r0w * 520 + kk * 32, 520);
#pragma unroll
      for (int t = 0; t < 4; ++t)
        acc[t] = wmma_bf16(a, frag_b(WoT + (cg * 64 + t * 16) * HD_ + kk * 32, HD_), acc[t]);
    }
#pragma unroll
    for (int t = 0; t < 4; ++t) {
      int cc = cg * 64 + t * 16 + (lane & 15);
      float bb = bo[cc];
#pragma unroll
      for (int i = 0; i < 8; ++i) {
        long long row = tile0 + r0w + ((lane >> 4) << 3) + i;
        out[row * C_ + cc] = acc[t][i] + bb;
      }
    }
  }
}

// ---------------- launcher ----------------
extern "C" void kernel_launch(void* const* d_in, const int* in_sizes, int n_in,
                              void* d_out, int out_size, void* d_ws, size_t ws_size,
                              hipStream_t stream) {
  (void)in_sizes; (void)n_in; (void)out_size; (void)ws_size;
  const float* x      = (const float*)d_in[0];
  const float* sonata = (const float*)d_in[1];
  const float* temp   = (const float*)d_in[2];
  const float* Wx  = (const float*)d_in[3];  const float* bx  = (const float*)d_in[4];
  const float* Wfx = (const float*)d_in[5];  const float* bfx = (const float*)d_in[6];
  const float* Wsl = (const float*)d_in[7];  const float* bsl = (const float*)d_in[8];
  const float* Wq  = (const float*)d_in[9];  const float* bq  = (const float*)d_in[10];
  const float* Wk  = (const float*)d_in[11]; const float* bk  = (const float*)d_in[12];
  const float* Wv  = (const float*)d_in[13]; const float* bv  = (const float*)d_in[14];
  const float* Wsp = (const float*)d_in[15]; const float* bsp = (const float*)d_in[16];
  const float* Wck = (const float*)d_in[17]; const float* bck = (const float*)d_in[18];
  const float* Wcv = (const float*)d_in[19]; const float* bcv = (const float*)d_in[20];
  const float* Wo  = (const float*)d_in[21]; const float* bo  = (const float*)d_in[22];
  float* out = (float*)d_out;

  char* wsp = (char*)d_ws;
  auto take = [&](size_t bytes) { char* p = wsp; wsp += (bytes + 255) & ~(size_t)255; return p; };
  __bf16* fx_ws = (__bf16*)take((size_t)B_ * N_ * HD_ * 2);
  __bf16* w_ws  = (__bf16*)take((size_t)B_ * H_ * N_ * G_ * 2);
  __bf16* sf_ws = (__bf16*)take((size_t)B_ * M_ * HD_ * 2);
  float*  stok  = (float*) take((size_t)B_ * H_ * G_ * D_ * 4);
  float*  snorm = (float*) take((size_t)B_ * H_ * G_ * 4);
  __bf16* osT   = (__bf16*)take((size_t)B_ * H_ * D_ * G_ * 2);
  __bf16* WxT   = (__bf16*)take((size_t)HD_ * C_ * 2);
  __bf16* WfxT  = (__bf16*)take((size_t)HD_ * C_ * 2);
  __bf16* WslT  = (__bf16*)take((size_t)G_ * D_ * 2);
  __bf16* WqT   = (__bf16*)take((size_t)D_ * D_ * 2);
  __bf16* WkT   = (__bf16*)take((size_t)D_ * D_ * 2);
  __bf16* WvT   = (__bf16*)take((size_t)D_ * D_ * 2);
  __bf16* WspT  = (__bf16*)take((size_t)HD_ * SD_ * 2);
  __bf16* WckT  = (__bf16*)take((size_t)D_ * D_ * 2);
  __bf16* WcvT  = (__bf16*)take((size_t)D_ * D_ * 2);
  __bf16* WoT   = (__bf16*)take((size_t)C_ * HD_ * 2);

  auto tr = [&](const float* W, __bf16* WT, int K, int Nn) {
    int total = K * Nn;
    k_transpose_bf16<<<(total + 255) / 256, 256, 0, stream>>>(W, WT, K, Nn);
  };
  tr(Wx, WxT, C_, HD_);   tr(Wfx, WfxT, C_, HD_); tr(Wsl, WslT, D_, G_);
  tr(Wq, WqT, D_, D_);    tr(Wk, WkT, D_, D_);    tr(Wv, WvT, D_, D_);
  tr(Wsp, WspT, SD_, HD_); tr(Wck, WckT, D_, D_); tr(Wcv, WcvT, D_, D_);
  tr(Wo, WoT, HD_, C_);
  k_zero_f32<<<(B_ * H_ * G_ * D_ + 255) / 256, 256, 0, stream>>>(stok, B_ * H_ * G_ * D_);
  k_zero_f32<<<(B_ * H_ * G_ + 255) / 256, 256, 0, stream>>>(snorm, B_ * H_ * G_);

  k_proj<<<B_ * N_ / 64, 128, 0, stream>>>(x, temp, WxT, bx, WfxT, bfx, WslT, bsl, fx_ws, w_ws);
  k_sonata<<<B_ * M_ / 64, 128, 0, stream>>>(sonata, WspT, bsp, sf_ws);
  k_pool<<<dim3(B_ * H_, 8), 128, 0, stream>>>(w_ws, fx_ws, stok, snorm);
  k_attn<<<B_ * H_, 128, 0, stream>>>(stok, snorm, sf_ws, WqT, bq, WkT, bk, WvT, bv,
                                      WckT, bck, WcvT, bcv, osT);
  k_unpool_out<<<B_ * N_ / 64, 128, 0, stream>>>(w_ws, osT, WoT, bo, out);
}
